// GATdecoderlayer_19164144075170
// MI455X (gfx1250) — matrix operand order repaired and checked by reading, not compiled, not run on hardware
//
#include <hip/hip_runtime.h>
#include <hip/hip_bf16.h>
#include <math.h>

typedef __attribute__((ext_vector_type(16))) __bf16 v16bf;
typedef __attribute__((ext_vector_type(8)))  float  v8f;
typedef __attribute__((ext_vector_type(4)))  float  v4f;

constexpr int kB   = 4;
constexpr int kN   = 2048;
constexpr int kIn  = 256;   // IN_DIM
constexpr int kH   = 4;     // HEADS
constexpr int kOut = 64;    // OUT_DIM per head (kH*kOut == kIn)

#define WMMA_BF16(A, Bt, C) \
  __builtin_amdgcn_wmma_f32_16x16x32_bf16(false, (A), false, (Bt), (short)0, (C), false, false)

// Build the 16x32 bf16 A-tile fragment for this lane from two contiguous
// 8-float runs (ISA 7.12.2 16-bit A layout: elems 0..7 -> K=g*8+0..7,
// elems 8..15 -> K=16+g*8+0..7, where g = lane>>4).
__device__ inline v16bf make_a(const v4f r0, const v4f r1,
                               const v4f r2, const v4f r3) {
  v16bf a;
  a[0]=(__bf16)r0[0];  a[1]=(__bf16)r0[1];  a[2]=(__bf16)r0[2];  a[3]=(__bf16)r0[3];
  a[4]=(__bf16)r1[0];  a[5]=(__bf16)r1[1];  a[6]=(__bf16)r1[2];  a[7]=(__bf16)r1[3];
  a[8]=(__bf16)r2[0];  a[9]=(__bf16)r2[1];  a[10]=(__bf16)r2[2]; a[11]=(__bf16)r2[3];
  a[12]=(__bf16)r3[0]; a[13]=(__bf16)r3[1]; a[14]=(__bf16)r3[2]; a[15]=(__bf16)r3[3];
  return a;
}

// ---------------------------------------------------------------------------
// Kernel 0: convert W -> bf16 (same layout) and Hw -> bf16 transposed
// (HwT[k][n] = Hw[n][k]) so gate GEMM's B operand is k-major contiguous.
// ---------------------------------------------------------------------------
__global__ void gat_prep_kernel(const float* __restrict__ W,
                                const float* __restrict__ Hw,
                                __bf16* __restrict__ Wb,
                                __bf16* __restrict__ HwT) {
  int i = blockIdx.x * blockDim.x + threadIdx.x;   // 0..65535
  Wb[i] = (__bf16)W[i];
  int k = i >> 8, n = i & 255;
  HwT[k * kIn + n] = (__bf16)Hw[n * kIn + k];
}

// ---------------------------------------------------------------------------
// Kernel 1: h[b,head,n,o] = feat_in[b,n,:] @ W[head]  (bf16 out, k-major rows)
// Block = 128 threads = 4 waves; wave w handles head w. Block -> (b, m-tile).
// ---------------------------------------------------------------------------
__global__ void gat_h_kernel(const float* __restrict__ feat,
                             const __bf16* __restrict__ Wb,
                             __bf16* __restrict__ hbuf) {
  const int lane = threadIdx.x & 31;
  const int head = threadIdx.x >> 5;
  const int b    = blockIdx.x / (kN / 16);
  const int m0   = (blockIdx.x % (kN / 16)) * 16;

  const float*  arow = feat + ((size_t)b * kN + (m0 + (lane & 15))) * kIn;
  const __bf16* wv   = Wb + (size_t)head * kIn * kOut;
  const int aoff = (lane >> 4) * 8;

  v8f c0 = {}, c1 = {}, c2 = {}, c3 = {};
  for (int k0 = 0; k0 < kIn; k0 += 32) {
    const v4f* ap = (const v4f*)(arow + k0 + aoff);
    v16bf a = make_a(ap[0], ap[1], ap[4], ap[5]);
    const __bf16* bcol = wv + (size_t)k0 * kOut + (lane & 15);
    v16bf b0, b1, b2, b3;
    for (int s = 0; s < 16; s++) {
      const __bf16* p = bcol + (size_t)((lane >> 4) * 8 + (s & 7) + (s >> 3) * 16) * kOut;
      b0[s] = p[0]; b1[s] = p[16]; b2[s] = p[32]; b3[s] = p[48];
    }
    c0 = WMMA_BF16(a, b0, c0);
    c1 = WMMA_BF16(a, b1, c1);
    c2 = WMMA_BF16(a, b2, c2);
    c3 = WMMA_BF16(a, b3, c3);
  }

  __bf16* hout = hbuf + (size_t)(b * kH + head) * kN * kOut;
  const int rbase = (lane >> 4) * 8, cb = lane & 15;
#pragma unroll
  for (int r = 0; r < 8; r++) {
    size_t ro = (size_t)(m0 + r + rbase) * kOut;
    hout[ro +  0 + cb] = (__bf16)c0[r];
    hout[ro + 16 + cb] = (__bf16)c1[r];
    hout[ro + 32 + cb] = (__bf16)c2[r];
    hout[ro + 48 + cb] = (__bf16)c3[r];
  }
}

// ---------------------------------------------------------------------------
// Kernel 2: gate[b,n,c] = sigmoid(feat_in[b,n,:] @ Hw^T + Hb)  (f32 out)
// Block = 4 waves; wave w handles output cols w*64..w*64+63. Block -> (b, m-tile).
// ---------------------------------------------------------------------------
__global__ void gat_gate_kernel(const float* __restrict__ feat,
                                const __bf16* __restrict__ HwT,
                                const float* __restrict__ Hb,
                                float* __restrict__ gate) {
  const int lane = threadIdx.x & 31;
  const int w    = threadIdx.x >> 5;
  const int b    = blockIdx.x / (kN / 16);
  const int m0   = (blockIdx.x % (kN / 16)) * 16;

  const float* arow = feat + ((size_t)b * kN + (m0 + (lane & 15))) * kIn;
  const int aoff = (lane >> 4) * 8;

  v8f acc[4] = {};
  for (int k0 = 0; k0 < kIn; k0 += 32) {
    const v4f* ap = (const v4f*)(arow + k0 + aoff);
    v16bf a = make_a(ap[0], ap[1], ap[4], ap[5]);
    const __bf16* bcol = HwT + (size_t)k0 * kIn + w * 64 + (lane & 15);
#pragma unroll
    for (int t = 0; t < 4; t++) {
      v16bf bt;
      for (int s = 0; s < 16; s++)
        bt[s] = bcol[(size_t)((lane >> 4) * 8 + (s & 7) + (s >> 3) * 16) * kIn + t * 16];
      acc[t] = WMMA_BF16(a, bt, acc[t]);
    }
  }

  const int rbase = (lane >> 4) * 8, cb = lane & 15;
#pragma unroll
  for (int t = 0; t < 4; t++) {
    int ch = w * 64 + t * 16 + cb;
    float bias = Hb[ch];
#pragma unroll
    for (int r = 0; r < 8; r++) {
      int rr = m0 + r + rbase;
      float x = acc[t][r] + bias;
      gate[((size_t)b * kN + rr) * kIn + ch] = 1.f / (1.f + expf(-x));
    }
  }
}

// ---------------------------------------------------------------------------
// Kernel 3 (hot): per (b,head): 16-row m-tiles of  feat_out = IB @ h.
// K=2048 is SPLIT across two waves (khalf 0/1) to double memory-level
// parallelism for the HBM-bound IB stream; partial f32 accumulators are
// combined through LDS. Each wave:
//   - streams its K-half of A = IB rows (fp32, nontemporal), fused copy to
//     d_out's b_inv region (each element touched exactly once grid-wide),
//     converts to bf16 A-fragments
//   - WMMA against h (bf16, L2-resident)
// khalf==0 waves then run the epilogue: +bias, ELU, sigmoid-gate blend.
// Block = 128 threads = 4 waves = 2 m-tiles x 2 k-halves. Grid = 16*64 blocks.
// ---------------------------------------------------------------------------
__global__ void gat_main_kernel(const float* __restrict__ IB,
                                const __bf16* __restrict__ hbuf,
                                const float* __restrict__ bias,
                                const float* __restrict__ feat,
                                const float* __restrict__ gate,
                                float* __restrict__ out,
                                float* __restrict__ ibcopy) {
  __shared__ float red[2 * 4 * 8 * 32];             // khalf-1 partials, 8 KB

  const int lane  = threadIdx.x & 31;
  const int w     = threadIdx.x >> 5;
  const int mtl   = w >> 1;                         // m-tile within block: 0/1
  const int khalf = w & 1;                          // K half: 0/1
  const int bh    = blockIdx.x >> 6;                // 0..15 = b*4+head
  const int m0    = (((blockIdx.x & 63) << 1) + mtl) * 16;
  const int b     = bh >> 2;
  const int head  = bh & 3;

  const float*  ibb = IB + (size_t)bh * kN * kN;
  float*        cpb = ibcopy + (size_t)bh * kN * kN;
  const __bf16* hb  = hbuf + (size_t)bh * kN * kOut;

  const int arowi = m0 + (lane & 15);
  const float* arow = ibb + (size_t)arowi * kN;
  float*       crow = cpb + (size_t)arowi * kN;
  const int aoff = (lane >> 4) * 8;
  const int kbeg = khalf * (kN / 2);
  const int kend = kbeg + (kN / 2);

  v8f acc[4] = {};
  for (int k0 = kbeg; k0 < kend; k0 += 32) {
    const v4f* ap = (const v4f*)(arow + k0 + aoff);
    v4f r0 = __builtin_nontemporal_load(ap + 0);
    v4f r1 = __builtin_nontemporal_load(ap + 1);
    v4f r2 = __builtin_nontemporal_load(ap + 4);
    v4f r3 = __builtin_nontemporal_load(ap + 5);

    // fused b_inv copy (each IB element stored exactly once across the grid)
    v4f* cp = (v4f*)(crow + k0 + aoff);
    __builtin_nontemporal_store(r0, cp + 0);
    __builtin_nontemporal_store(r1, cp + 1);
    __builtin_nontemporal_store(r2, cp + 4);
    __builtin_nontemporal_store(r3, cp + 5);

    v16bf a = make_a(r0, r1, r2, r3);

    const __bf16* bcol = hb + (size_t)k0 * kOut + (lane & 15);  // lane&15 = N col of B
    v16bf b0, b1, b2, b3;
    for (int s = 0; s < 16; s++) {
      const __bf16* p = bcol + (size_t)((lane >> 4) * 8 + (s & 7) + (s >> 3) * 16) * kOut;
      b0[s] = p[0]; b1[s] = p[16]; b2[s] = p[32]; b3[s] = p[48];
    }
    acc[0] = WMMA_BF16(a, b0, acc[0]);
    acc[1] = WMMA_BF16(a, b1, acc[1]);
    acc[2] = WMMA_BF16(a, b2, acc[2]);
    acc[3] = WMMA_BF16(a, b3, acc[3]);
  }

  // Cross-wave K reduction: khalf==1 parks its partials in LDS.
  if (khalf == 1) {
#pragma unroll
    for (int t = 0; t < 4; t++)
#pragma unroll
      for (int r = 0; r < 8; r++)
        red[((mtl * 4 + t) * 8 + r) * 32 + lane] = acc[t][r];
  }
  __syncthreads();
  if (khalf == 1) return;

#pragma unroll
  for (int t = 0; t < 4; t++)
#pragma unroll
    for (int r = 0; r < 8; r++)
      acc[t][r] += red[((mtl * 4 + t) * 8 + r) * 32 + lane];

  // Epilogue: D layout is VGPR r, lane L -> M = r + (L>>4)*8, N = L&15.
  const int rbase = (lane >> 4) * 8, cb = lane & 15;
#pragma unroll
  for (int t = 0; t < 4; t++) {
    int o  = t * 16 + cb;
    int ch = head * kOut + o;
    float bv = bias[o];
#pragma unroll
    for (int r = 0; r < 8; r++) {
      int rr = m0 + r + rbase;
      float fo = acc[t][r] + bv;
      float e  = fo > 0.f ? fo : expm1f(fo);            // ELU(alpha=1)
      size_t idx = ((size_t)b * kN + rr) * kIn + ch;
      float g  = gate[idx];
      float fi = feat[idx];
      out[idx] = g * e + (1.f - g) * fi;
    }
  }
}

// ---------------------------------------------------------------------------
extern "C" void kernel_launch(void* const* d_in, const int* in_sizes, int n_in,
                              void* d_out, int out_size, void* d_ws, size_t ws_size,
                              hipStream_t stream) {
  const float* feat = (const float*)d_in[0];   // [4,2048,256]
  const float* IB   = (const float*)d_in[1];   // [4,4,2048,2048]
  const float* W    = (const float*)d_in[2];   // [4,256,64]
  const float* bias = (const float*)d_in[3];   // [64]
  const float* Hw   = (const float*)d_in[4];   // [256,256]
  const float* Hb   = (const float*)d_in[5];   // [256]

  float* outp   = (float*)d_out;                                 // [4,2048,256]
  float* ibcopy = outp + (size_t)kB * kN * kIn;                  // b_inv copy

  // workspace layout
  __bf16* hbuf = (__bf16*)d_ws;                                  // 4 MB
  float*  gate = (float*)((char*)d_ws + (4u << 20));             // 8 MB
  __bf16* Wb   = (__bf16*)((char*)d_ws + (12u << 20));           // 128 KB
  __bf16* HwT  = (__bf16*)((char*)d_ws + (12u << 20) + (128u << 10)); // 128 KB

  gat_prep_kernel<<<256, 256, 0, stream>>>(W, Hw, Wb, HwT);
  gat_h_kernel<<<kB * (kN / 16), 128, 0, stream>>>(feat, Wb, hbuf);
  gat_gate_kernel<<<kB * (kN / 16), 128, 0, stream>>>(feat, HwT, Hb, gate);
  gat_main_kernel<<<(kB * kH) * (kN / 16 / 2), 128, 0, stream>>>(
      IB, hbuf, bias, feat, gate, outp, ibcopy);
}
